// MultiHeadAttention_69879117906270
// MI455X (gfx1250) — matrix operand order repaired and checked by reading, not compiled, not run on hardware
//
#include <hip/hip_runtime.h>
#include <hip/hip_bf16.h>

// ---------------------------------------------------------------------------
// Multi-head attention forward for MI455X (gfx1250), wave32 + WMMA bf16.
//   out  = (softmax(QK^T/sqrt(dk)) V) Wo + bo      -> d_out[0 .. 4M)
//   attn = softmax scores                          -> d_out[4M .. 4M+64M)
// All GEMMs use v_wmma_f32_16x16x32_bf16 (f32 accumulate).
// Per-wave macro-tiles are 16x64: the A fragment is loaded once per k-step
// and fed to 4 WMMAs, maximizing wmma:vmem and minimizing HBM re-reads of
// the 268 MB attention tensor (the bandwidth-dominant stream at 23.3 TB/s).
// Workspace requirement: 48 MiB.
// ---------------------------------------------------------------------------

typedef __attribute__((ext_vector_type(16))) __bf16 v16bf;
typedef __attribute__((ext_vector_type(8)))  float  v8f;

constexpr int Bc = 4;      // batch
constexpr int Sc = 1024;   // sequence
constexpr int Dm = 1024;   // d_model
constexpr int Hc = 16;     // heads
constexpr int Dk = 64;     // head dim

// ---------------- WMMA helpers (layouts per CDNA5 ISA 7.12.2) --------------

__device__ __forceinline__ v8f wmma_bf16(v16bf a, v16bf b, v8f c) {
  return __builtin_amdgcn_wmma_f32_16x16x32_bf16(
      /*neg_a=*/false, a, /*neg_b=*/false, b,
      /*c_mod=*/(short)0, c, /*reuse_a=*/false, /*reuse_b=*/false);
}

// A-matrix 16x32 bf16, row-major source, row stride = lda elements.
// lane(0..15): M = lane; lanes 16..31 hold the upper K halves.
// a[0..7]  <- K = half*8 .. half*8+7 ; a[8..15] <- K = 16+half*8 .. +7
__device__ __forceinline__ v16bf load_a_bf16(const __bf16* base, int lda, int lane) {
  const int row = lane & 15, half = lane >> 4;
  const __bf16* p = base + (size_t)row * lda + half * 8;
  union { v16bf v; uint4 q[2]; } u;
  u.q[0] = *(const uint4*)(p);
  u.q[1] = *(const uint4*)(p + 16);
  return u.v;
}

// Same A layout, f32 source, converted to bf16 in registers.
__device__ __forceinline__ v16bf load_a_f32(const float* base, int lda, int lane) {
  const int row = lane & 15, half = lane >> 4;
  const float* p = base + (size_t)row * lda + half * 8;
  float4 x0 = *(const float4*)(p);
  float4 x1 = *(const float4*)(p + 4);
  float4 y0 = *(const float4*)(p + 16);
  float4 y1 = *(const float4*)(p + 20);
  v16bf a;
  a[0]=(__bf16)x0.x; a[1]=(__bf16)x0.y; a[2]=(__bf16)x0.z; a[3]=(__bf16)x0.w;
  a[4]=(__bf16)x1.x; a[5]=(__bf16)x1.y; a[6]=(__bf16)x1.z; a[7]=(__bf16)x1.w;
  a[8]=(__bf16)y0.x; a[9]=(__bf16)y0.y; a[10]=(__bf16)y0.z; a[11]=(__bf16)y0.w;
  a[12]=(__bf16)y1.x; a[13]=(__bf16)y1.y; a[14]=(__bf16)y1.z; a[15]=(__bf16)y1.w;
  return a;
}

// B-matrix 32x16: source stored TRANSPOSED (Bt[n][k], row stride ldb); each
// lane (col n = lane&15, half = lane>>4) reads K = half*8..+7 and 16+half*8..+7.
__device__ __forceinline__ v16bf load_b_bt(const __bf16* base, int ldb, int lane) {
  const int n = lane & 15, half = lane >> 4;
  const __bf16* p = base + (size_t)n * ldb + half * 8;
  union { v16bf v; uint4 q[2]; } u;
  u.q[0] = *(const uint4*)(p);
  u.q[1] = *(const uint4*)(p + 16);
  return u.v;
}

__device__ __forceinline__ float wave_max(float v) {
  for (int m = 16; m > 0; m >>= 1) v = fmaxf(v, __shfl_xor(v, m, 32));
  return v;
}
__device__ __forceinline__ float wave_sum(float v) {
  for (int m = 16; m > 0; m >>= 1) v += __shfl_xor(v, m, 32);
  return v;
}

// ---------------- small prep kernels ---------------------------------------

// Wt[n*1024 + k] = bf16(W[k*1024 + n])   (1M elements)
__global__ __launch_bounds__(256) void wcvt_kernel(const float* __restrict__ W,
                                                   __bf16* __restrict__ Wt) {
  int idx = blockIdx.x * 256 + threadIdx.x;
  int k = idx >> 10, n = idx & 1023;
  Wt[(size_t)n * 1024 + k] = (__bf16)W[idx];
}

// Vt[b,h,d,s] = Vb[b,s,h*64+d]   (4M bf16 elements)
__global__ __launch_bounds__(256) void vtrans_kernel(const __bf16* __restrict__ Vb,
                                                     __bf16* __restrict__ Vt) {
  int idx = blockIdx.x * 256 + threadIdx.x;
  int dm = idx & 1023, s = (idx >> 10) & 1023, b = idx >> 20;
  int h = dm >> 6, d = dm & 63;
  Vt[((size_t)((b * Hc + h) * Dk + d)) * Sc + s] = Vb[idx];
}

// ---------------- GEMM: C[4096,1024] = A[4096,1024] * Bt^T + bias ----------
// 8 waves/block; each wave computes a 16x64 macro-tile (4 WMMA accumulators),
// reusing the A fragment across the 4 n-subtiles. K=1024 in 32 steps.
template <int A_F32, int OUT_BF16>
__global__ __launch_bounds__(256) void gemm16_kernel(const void* __restrict__ Ap,
                                                     const __bf16* __restrict__ Bt,
                                                     const float* __restrict__ bias,
                                                     void* __restrict__ Cp) {
  const int lane = threadIdx.x & 31;
  const int wave = threadIdx.x >> 5;
  const int tile = blockIdx.x * 8 + wave;   // 4096 wave-tiles (256 m x 16 n-groups)
  const int m0 = (tile >> 4) << 4;          // 16 n-groups of 64 per m-row
  const int n0 = (tile & 15) << 6;

  v8f acc[4] = {};
  for (int k0 = 0; k0 < 1024; k0 += 32) {
    v16bf a;
    if (A_F32) a = load_a_f32((const float*)Ap + (size_t)m0 * 1024 + k0, 1024, lane);
    else       a = load_a_bf16((const __bf16*)Ap + (size_t)m0 * 1024 + k0, 1024, lane);
#pragma unroll
    for (int j = 0; j < 4; ++j) {
      v16bf bfr = load_b_bt(Bt + (size_t)(n0 + j * 16) * 1024 + k0, 1024, lane);
      acc[j] = wmma_bf16(a, bfr, acc[j]);
    }
  }
  const int col = lane & 15, rb = (lane >> 4) * 8;
#pragma unroll
  for (int j = 0; j < 4; ++j) {
    const int nj = n0 + j * 16 + col;
    const float bv = bias[nj];
#pragma unroll
    for (int r = 0; r < 8; ++r) {
      float v = acc[j][r] + bv;
      size_t off = (size_t)(m0 + rb + r) * 1024 + nj;
      if (OUT_BF16) ((__bf16*)Cp)[off] = (__bf16)v;
      else          ((float*)Cp)[off] = v;
    }
  }
}

// ---------------- scores + softmax -----------------------------------------
// One block per (b, h, 16-query tile). 8 waves compute 16x1024 scores via
// WMMA into LDS (64 KB), then wave-parallel softmax -> attn (f32, d_out).
// Q fragments are loaded once per wave and reused across all 8 key tiles.
__global__ __launch_bounds__(256) void scores_softmax_kernel(const __bf16* __restrict__ Qb,
                                                             const __bf16* __restrict__ Kb,
                                                             float* __restrict__ attn) {
  extern __shared__ float sc[];             // 16 x 1024 f32
  const int lane = threadIdx.x & 31;
  const int wave = threadIdx.x >> 5;
  const int qt = blockIdx.x & 63;
  const int bh = blockIdx.x >> 6;
  const int h = bh & 15, b = bh >> 4;
  const int q0 = qt * 16;
  const int col = lane & 15, rb = (lane >> 4) * 8;

  const __bf16* Qbase = Qb + ((size_t)(b * Sc + q0)) * Dm + h * Dk;
  const v16bf aq0 = load_a_bf16(Qbase, 1024, lane);        // d_k 0..31
  const v16bf aq1 = load_a_bf16(Qbase + 32, 1024, lane);   // d_k 32..63

#pragma unroll
  for (int t = 0; t < 8; ++t) {
    const int n0 = wave * 128 + t * 16;                    // key tile
    const __bf16* Kbase = Kb + ((size_t)(b * Sc + n0)) * Dm + h * Dk;
    v8f acc = {};
    acc = wmma_bf16(aq0, load_b_bt(Kbase, 1024, lane), acc);
    acc = wmma_bf16(aq1, load_b_bt(Kbase + 32, 1024, lane), acc);
#pragma unroll
    for (int r = 0; r < 8; ++r)
      sc[(rb + r) * 1024 + n0 + col] = acc[r] * 0.125f;    // 1/sqrt(64)
  }
  __syncthreads();

  // softmax: each wave owns 2 of the 16 query rows
#pragma unroll
  for (int rr = 0; rr < 2; ++rr) {
    const int row = wave * 2 + rr;
    float* srow = sc + row * 1024;
    float m = -3.402823e38f;
    for (int j = lane; j < 1024; j += 32) m = fmaxf(m, srow[j]);
    m = wave_max(m);
    float s = 0.f;
    for (int j = lane; j < 1024; j += 32) {
      float e = __expf(srow[j] - m);
      srow[j] = e;
      s += e;
    }
    s = wave_sum(s);
    const float inv = 1.0f / s;
    float* orow = attn + ((size_t)(b * Hc + h) * Sc + q0 + row) * Sc;
    for (int j = lane; j < 1024; j += 32) orow[j] = srow[j] * inv;
  }
}

// ---------------- context: ctx[b,q,h*64+d] = attn[b,h,q,:] . V[b,:,h*64+d] --
// One wave owns a full (16 queries x 64 d) tile: the attn fragment (the
// HBM-dominant 268 MB stream) is loaded ONCE per k-step and reused by 4 WMMAs.
// 8 waves/block -> grid = B*H*(S/16)/8 = 512 blocks.
__global__ __launch_bounds__(256) void ctx_kernel(const float* __restrict__ attn,
                                                  const __bf16* __restrict__ Vt,
                                                  __bf16* __restrict__ ctx) {
  const int lane = threadIdx.x & 31;
  const int wave = threadIdx.x >> 5;
  const int tile = blockIdx.x * 8 + wave;   // 4096 (b,h,qt) tiles
  const int qt = tile & 63;
  const int bh = tile >> 6;
  const int h = bh & 15, b = bh >> 4;
  const int q0 = qt * 16;

  const float* Abase = attn + ((size_t)bh * Sc + q0) * Sc;
  const __bf16* Bbase = Vt + (size_t)bh * Dk * Sc;

  v8f acc[4] = {};
  for (int k0 = 0; k0 < 1024; k0 += 32) {
    v16bf a = load_a_f32(Abase + k0, 1024, lane);
#pragma unroll
    for (int j = 0; j < 4; ++j) {
      v16bf bfr = load_b_bt(Bbase + (size_t)(j * 16) * Sc + k0, 1024, lane);
      acc[j] = wmma_bf16(a, bfr, acc[j]);
    }
  }
  const int col = lane & 15, rb = (lane >> 4) * 8;
#pragma unroll
  for (int j = 0; j < 4; ++j)
#pragma unroll
    for (int r = 0; r < 8; ++r)
      ctx[(size_t)(b * Sc + q0 + rb + r) * Dm + h * Dk + j * 16 + col] =
          (__bf16)acc[j][r];
}

// ---------------- launch ----------------------------------------------------

extern "C" void kernel_launch(void* const* d_in, const int* in_sizes, int n_in,
                              void* d_out, int out_size, void* d_ws, size_t ws_size,
                              hipStream_t stream) {
  const float* q  = (const float*)d_in[0];
  const float* k  = (const float*)d_in[1];
  const float* v  = (const float*)d_in[2];
  const float* Wq = (const float*)d_in[3];
  const float* bq = (const float*)d_in[4];
  const float* Wk = (const float*)d_in[5];
  const float* bk = (const float*)d_in[6];
  const float* Wv = (const float*)d_in[7];
  const float* bv = (const float*)d_in[8];
  const float* Wo = (const float*)d_in[9];
  const float* bo = (const float*)d_in[10];

  // workspace layout (48 MiB total)
  const size_t MB = 1024 * 1024;
  char* ws = (char*)d_ws;
  __bf16* Wqt = (__bf16*)(ws + 0 * MB);     // 2 MiB each (1M bf16)
  __bf16* Wkt = (__bf16*)(ws + 2 * MB);
  __bf16* Wvt = (__bf16*)(ws + 4 * MB);
  __bf16* Wot = (__bf16*)(ws + 6 * MB);
  __bf16* Qb  = (__bf16*)(ws + 8 * MB);     // 8 MiB each (4M bf16)
  __bf16* Kb  = (__bf16*)(ws + 16 * MB);
  __bf16* Vb  = (__bf16*)(ws + 24 * MB);
  __bf16* Vt  = (__bf16*)(ws + 32 * MB);
  __bf16* Ctx = (__bf16*)(ws + 40 * MB);

  float* outO  = (float*)d_out;                       // [4,1024,1024]
  float* attnO = outO + (size_t)Bc * Sc * Dm;         // [4,16,1024,1024]

  // 1) weight convert+transpose (f32 -> bf16 W^T)
  wcvt_kernel<<<4096, 256, 0, stream>>>(Wq, Wqt);
  wcvt_kernel<<<4096, 256, 0, stream>>>(Wk, Wkt);
  wcvt_kernel<<<4096, 256, 0, stream>>>(Wv, Wvt);
  wcvt_kernel<<<4096, 256, 0, stream>>>(Wo, Wot);

  // 2) projections: [4096,1024] x [1024,1024] + bias -> bf16 (16x64/wave)
  gemm16_kernel<1, 1><<<512, 256, 0, stream>>>(q, Wqt, bq, Qb);
  gemm16_kernel<1, 1><<<512, 256, 0, stream>>>(k, Wkt, bk, Kb);
  gemm16_kernel<1, 1><<<512, 256, 0, stream>>>(v, Wvt, bv, Vb);

  // 3) V transpose to [b,h,d,s] for contiguous B-fragments
  vtrans_kernel<<<16384, 256, 0, stream>>>(Vb, Vt);

  // 4) scores + softmax -> attn output (f32)
  scores_softmax_kernel<<<Bc * Hc * (Sc / 16), 256, 16 * 1024 * sizeof(float), stream>>>(
      Qb, Kb, attnO);

  // 5) context = attn @ V -> bf16 (one wave per 16x64 tile, attn read once)
  ctx_kernel<<<512, 256, 0, stream>>>(attnO, Vt, Ctx);

  // 6) output projection -> f32 out
  gemm16_kernel<0, 0><<<512, 256, 0, stream>>>(Ctx, Wot, bo, outO);
}
